// CausalMHA_61993557950513
// MI455X (gfx1250) — matrix operand order repaired and checked by reading, not compiled, not run on hardware
//
#include <hip/hip_runtime.h>

// ---------------- problem constants ----------------
#define B_   2
#define S_   2048
#define D_   2048
#define H_   16
#define DK_  128
#define DH_  64          // DK/2 (rope pairs)

typedef __attribute__((ext_vector_type(16))) __bf16 v16bf;
typedef __attribute__((ext_vector_type(8)))  float  v8f;
typedef __attribute__((ext_vector_type(4)))  int    i32x4;
typedef __attribute__((address_space(1))) i32x4 gi32x4;   // global int4
typedef __attribute__((address_space(3))) i32x4 li32x4;   // LDS int4

__device__ __forceinline__ v8f vzero() {
  v8f z;
#pragma unroll
  for (int i = 0; i < 8; ++i) z[i] = 0.f;
  return z;
}

__device__ __forceinline__ v8f wmma_bf16(v16bf a, v16bf b, v8f c) {
  // (neg_a, A, neg_b, B, c_mod, C, reuse_a, reuse_b)
  return __builtin_amdgcn_wmma_f32_16x16x32_bf16(false, a, false, b, (short)0, c,
                                                 false, false);
}

// A fragment, 16x32 bf16 (M x K). Lanes 0-15 row M=lane, K = 0..15 contiguous;
// lanes 16-31 same rows, K = 16..31.
__device__ __forceinline__ v16bf load_a_frag(const __bf16* t, int ld, int kbase) {
  const int lane = threadIdx.x & 31;
  const __bf16* row = t + (lane & 15) * ld + kbase + ((lane >> 4) << 4);
  v16bf f;
#pragma unroll
  for (int i = 0; i < 8; ++i) f[i] = row[i];
#pragma unroll
  for (int i = 0; i < 8; ++i) f[8 + i] = row[8 + i];
  return f;
}

// B fragment, 32x16 bf16 (K x N), tile stored N-major: t[n*ld + k].
// Lanes 0-15 col N=lane hold K=0..15 contiguous; lanes 16-31 K=16..31.
__device__ __forceinline__ v16bf load_b_frag(const __bf16* t, int ld, int kbase) {
  const int lane = threadIdx.x & 31;
  const __bf16* row = t + (lane & 15) * ld + kbase + ((lane >> 4) << 4);
  v16bf f;
#pragma unroll
  for (int i = 0; i < 16; ++i) f[i] = row[i];
  return f;
}

// ---------------- gfx1250 async global->LDS copy ----------------
#if defined(__has_builtin)
#if __has_builtin(__builtin_amdgcn_global_load_async_to_lds_b128)
#define HAVE_ASYNC_COPY 1
#endif
#endif

__device__ __forceinline__ void cp_b128(const __bf16* g, __bf16* l) {
#ifdef HAVE_ASYNC_COPY
  __builtin_amdgcn_global_load_async_to_lds_b128((gi32x4*)g, (li32x4*)l, 0, 0);
#else
  *(float4*)l = *(const float4*)g;     // 16B synchronous fallback
#endif
}

__device__ __forceinline__ void wait_async() {
#ifdef HAVE_ASYNC_COPY
#if __has_builtin(__builtin_amdgcn_s_wait_asynccnt)
  __builtin_amdgcn_s_wait_asynccnt(0);
#else
  asm volatile("s_wait_asynccnt 0" ::: "memory");
#endif
#endif
}

// pack a float4 into 4 bf16 at d
__device__ __forceinline__ void store_bf4(__bf16* d, float4 v) {
  d[0] = (__bf16)v.x; d[1] = (__bf16)v.y; d[2] = (__bf16)v.z; d[3] = (__bf16)v.w;
}

// ---------------- kernel 1: QKV projection + RoPE ----------------
// C[m, n] = sum_k x[m,k] * w_qkv[n,k] ; block tile 128x64, wave tile 32x64
#define GM 128
#define GN 64
#define GK 32
#define XSTR 48   // 96B rows, 16B aligned

__global__ __launch_bounds__(128) void qkv_rope_kernel(
    const float* __restrict__ x, const float* __restrict__ wqkv,
    const float* __restrict__ cosT, const float* __restrict__ sinT,
    __bf16* __restrict__ qw, __bf16* __restrict__ kw, __bf16* __restrict__ vw) {
  __shared__ __bf16 Xs[GM * XSTR];
  __shared__ __bf16 Ws[GN * XSTR];
  const int tid  = threadIdx.x;
  const int wave = tid >> 5;
  const int lane = tid & 31;
  const int m0 = blockIdx.y * GM;
  const int n0 = blockIdx.x * GN;

  // staging decomposition (constant per thread): 16 rows x 8 float4 per iter
  const int rx = tid >> 3;          // 0..15
  const int cx = (tid & 7) << 2;    // 0..28
  const float* xg = x    + (size_t)(m0 + rx) * D_ + cx;
  const float* wg = wqkv + (size_t)(n0 + rx) * D_ + cx;
  __bf16* xs = Xs + rx * XSTR + cx;
  __bf16* ws = Ws + rx * XSTR + cx;

  v8f acc[2][4];
#pragma unroll
  for (int mi = 0; mi < 2; ++mi)
#pragma unroll
    for (int j = 0; j < 4; ++j) acc[mi][j] = vzero();

  for (int k0 = 0; k0 < D_; k0 += GK) {
    __syncthreads();
#pragma unroll
    for (int it = 0; it < 8; ++it)                       // GM = 8*16 rows
      store_bf4(xs + it * 16 * XSTR,
                *(const float4*)(xg + (size_t)it * 16 * D_ + k0));
#pragma unroll
    for (int it = 0; it < 4; ++it)                       // GN = 4*16 rows
      store_bf4(ws + it * 16 * XSTR,
                *(const float4*)(wg + (size_t)it * 16 * D_ + k0));
    __syncthreads();
    v16bf a0 = load_a_frag(Xs + (wave * 32 +  0) * XSTR, XSTR, 0);
    v16bf a1 = load_a_frag(Xs + (wave * 32 + 16) * XSTR, XSTR, 0);
#pragma unroll
    for (int j = 0; j < 4; ++j) {
      v16bf b = load_b_frag(Ws + j * 16 * XSTR, XSTR, 0);
      acc[0][j] = wmma_bf16(a0, b, acc[0][j]);
      acc[1][j] = wmma_bf16(a1, b, acc[1][j]);
    }
  }

  // epilogue: RoPE for q/k, store bf16 into [B,H,S,DK] workspaces
  const int half  = lane >> 4;
  const int colLo = lane & 15;
#pragma unroll
  for (int mi = 0; mi < 2; ++mi) {
#pragma unroll
    for (int j = 0; j < 4; ++j) {
      const int n      = n0 + j * 16 + colLo;
      const int region = n >> 11;        // 0=q 1=k 2=v (uniform per tile)
      const int nd     = n & (D_ - 1);
      const int h      = nd >> 7;
      const int dk     = nd & (DK_ - 1);
      const int i2     = dk >> 1;
      const bool even  = (dk & 1) == 0;
#pragma unroll
      for (int r = 0; r < 8; ++r) {
        const int m  = m0 + wave * 32 + mi * 16 + r + (half << 3);
        const int bb = m >> 11;          // / S_
        const int s  = m & (S_ - 1);
        float xv = acc[mi][j][r];
        float ov = __shfl_xor(xv, 1);    // rope partner column
        float res = xv;
        if (region < 2) {
          float c  = cosT[s * DH_ + i2];
          float sn = sinT[s * DH_ + i2];
          res = even ? (xv * c - ov * sn) : (ov * sn + xv * c);
        }
        const size_t idx = ((size_t)(bb * H_ + h) * S_ + s) * DK_ + dk;
        if (region == 0)      qw[idx] = (__bf16)res;
        else if (region == 1) kw[idx] = (__bf16)res;
        else                  vw[idx] = (__bf16)res;
      }
    }
  }
}

// ---------------- kernel 2: causal flash attention ----------------
// double-buffered K/V tiles: async prefetch of block t+1 overlaps compute of t
#define QSTR 144  // 288B rows, 16B aligned
#define KSTR 144
#define VSTR 72   // 144B rows
#define PSTR 72

__global__ __launch_bounds__(128) void attn_kernel(
    const __bf16* __restrict__ qw, const __bf16* __restrict__ kw,
    const __bf16* __restrict__ vw, __bf16* __restrict__ ow) {
  __shared__ __bf16 Qs[64 * QSTR];
  __shared__ __bf16 Ks[2][64 * KSTR];
  __shared__ __bf16 Vt[2][DK_ * VSTR];     // transposed: Vt[buf][dk][krow]
  __shared__ __bf16 Ps[4 * 16 * PSTR];     // per-wave P tile (16 x 64)

  const int tid   = threadIdx.x;
  const int wave  = tid >> 5;
  const int lane  = tid & 31;
  const int half  = lane >> 4;
  const int colLo = lane & 15;
  const int qblk  = blockIdx.x;
  const int bh    = blockIdx.y;            // b*H + h
  const int q0    = qblk * 64;

  const __bf16* qp = qw + (size_t)bh * S_ * DK_;
  const __bf16* kp = kw + (size_t)bh * S_ * DK_;
  const __bf16* vp = vw + (size_t)bh * S_ * DK_;

  // staging decomposition: 8 rows x 16 b128 per iter
  const int rq = tid >> 4;          // 0..7
  const int cq = (tid & 15) << 3;   // 0..120

  // prologue: async Q tile + async K block 0, sync-transposed V block 0
#pragma unroll
  for (int it = 0; it < 8; ++it) {
    const int r = rq + it * 8;
    cp_b128(qp + (size_t)(q0 + r) * DK_ + cq, Qs + r * QSTR + cq);
    cp_b128(kp + (size_t)r * DK_ + cq, Ks[0] + r * KSTR + cq);
  }
#pragma unroll
  for (int it = 0; it < 8; ++it) {
    const int r = rq + it * 8;
    const float4 vv = *(const float4*)(vp + (size_t)r * DK_ + cq);
    const __bf16* v8p = (const __bf16*)&vv;
#pragma unroll
    for (int i = 0; i < 8; ++i)
      Vt[0][(cq + i) * VSTR + r] = v8p[i];
  }
  wait_async();
  __syncthreads();

  // hoist Q fragments (constant across all kv blocks)
  v16bf qfrag[4];
#pragma unroll
  for (int kc = 0; kc < 4; ++kc)
    qfrag[kc] = load_a_frag(Qs + wave * 16 * QSTR, QSTR, kc * 32);

  v8f o_acc[8];
#pragma unroll
  for (int j = 0; j < 8; ++j) o_acc[j] = vzero();
  float rmax[8], rsum[8];
#pragma unroll
  for (int r = 0; r < 8; ++r) { rmax[r] = -3.0e38f; rsum[r] = 0.f; }

  const float scale = 0.08838834764831845f;   // 1/sqrt(DK)
  __bf16* ps = Ps + wave * 16 * PSTR;

  for (int t = 0; t <= qblk; ++t) {
    const int cur = t & 1;
    const int nxt = cur ^ 1;
    const int kv0 = t * 64;

    // prefetch next kv block into the other buffer (overlaps with compute)
    if (t < qblk) {
      const int kv1 = kv0 + 64;
#pragma unroll
      for (int it = 0; it < 8; ++it) {
        const int r = rq + it * 8;
        cp_b128(kp + (size_t)(kv1 + r) * DK_ + cq, Ks[nxt] + r * KSTR + cq);
      }
#pragma unroll
      for (int it = 0; it < 8; ++it) {
        const int r = rq + it * 8;
        const float4 vv = *(const float4*)(vp + (size_t)(kv1 + r) * DK_ + cq);
        const __bf16* v8p = (const __bf16*)&vv;
#pragma unroll
        for (int i = 0; i < 8; ++i)
          Vt[nxt][(cq + i) * VSTR + r] = v8p[i];
      }
    }

    // S = Q K^T : per wave 16x64, K-dim = DK = 4 chunks of 32
    v8f s_acc[4];
#pragma unroll
    for (int j = 0; j < 4; ++j) s_acc[j] = vzero();
#pragma unroll
    for (int kc = 0; kc < 4; ++kc) {
#pragma unroll
      for (int j = 0; j < 4; ++j) {
        v16bf b = load_b_frag(Ks[cur] + j * 16 * KSTR, KSTR, kc * 32);
        s_acc[j] = wmma_bf16(qfrag[kc], b, s_acc[j]);
      }
    }

    // scale + causal mask + online softmax (row = r + 8*half, cols across lanes)
#pragma unroll
    for (int r = 0; r < 8; ++r) {
      const int qrow = q0 + wave * 16 + r + (half << 3);
      float sv[4];
      float tmax = -3.0e38f;
#pragma unroll
      for (int j = 0; j < 4; ++j) {
        const int key = kv0 + j * 16 + colLo;
        float v = s_acc[j][r] * scale;
        if (key > qrow) v = -3.0e38f;
        sv[j] = v;
        tmax = fmaxf(tmax, v);
      }
#pragma unroll
      for (int off = 1; off < 16; off <<= 1)
        tmax = fmaxf(tmax, __shfl_xor(tmax, off));
      const float mnew = fmaxf(rmax[r], tmax);
      const float resc = __expf(rmax[r] - mnew);
      float psum = 0.f;
#pragma unroll
      for (int j = 0; j < 4; ++j) {
        float p = __expf(sv[j] - mnew);
        psum += p;
        ps[(r + (half << 3)) * PSTR + j * 16 + colLo] = (__bf16)p;
      }
#pragma unroll
      for (int off = 1; off < 16; off <<= 1)
        psum += __shfl_xor(psum, off);
      rsum[r] = rsum[r] * resc + psum;
      rmax[r] = mnew;
#pragma unroll
      for (int j = 0; j < 8; ++j) o_acc[j][r] *= resc;
    }

    // O += P V : P 16x64 (A frags from LDS), V 64x128 (B frags from Vt)
#pragma unroll
    for (int kc = 0; kc < 2; ++kc) {
      v16bf a = load_a_frag(ps, PSTR, kc * 32);
#pragma unroll
      for (int j = 0; j < 8; ++j) {
        v16bf b = load_b_frag(Vt[cur] + j * 16 * VSTR, VSTR, kc * 32);
        o_acc[j] = wmma_bf16(a, b, o_acc[j]);
      }
    }

    // next buffer complete + all waves done with current before swap
    wait_async();
    __syncthreads();
  }

  // normalize and store O as bf16 into [B, S, H*DK] for the out-projection
  const int bb = bh / H_;
  const int h  = bh % H_;
#pragma unroll
  for (int j = 0; j < 8; ++j) {
#pragma unroll
    for (int r = 0; r < 8; ++r) {
      const int s  = q0 + wave * 16 + r + (half << 3);
      const int dk = j * 16 + colLo;
      float v = o_acc[j][r] / rsum[r];
      ow[((size_t)(bb * S_ + s)) * D_ + h * DK_ + dk] = (__bf16)v;
    }
  }
}

// ---------------- kernel 3: output projection ----------------
// out[m, n] = sum_k o[m,k] * w_out[n,k] ; block tile 128x64, wave tile 32x64
__global__ __launch_bounds__(128) void outproj_kernel(
    const __bf16* __restrict__ ow, const float* __restrict__ wout,
    float* __restrict__ out) {
  __shared__ __bf16 Os[GM * XSTR];
  __shared__ __bf16 Ws[GN * XSTR];
  const int tid  = threadIdx.x;
  const int wave = tid >> 5;
  const int lane = tid & 31;
  const int m0 = blockIdx.y * GM;
  const int n0 = blockIdx.x * GN;

  // staging decomposition
  const int ro = tid >> 2;          // 0..31 (bf16 tile: 32 rows x 4 b128/row)
  const int co = (tid & 3) << 3;    // 0,8,16,24
  const int rx = tid >> 3;          // 0..15 (f32 tile: 16 rows x 8 float4/row)
  const int cx = (tid & 7) << 2;    // 0..28
  const __bf16* og = ow   + (size_t)(m0 + ro) * D_ + co;
  const float*  wg = wout + (size_t)(n0 + rx) * D_ + cx;
  __bf16* os = Os + ro * XSTR + co;
  __bf16* ws = Ws + rx * XSTR + cx;

  v8f acc[2][4];
#pragma unroll
  for (int mi = 0; mi < 2; ++mi)
#pragma unroll
    for (int j = 0; j < 4; ++j) acc[mi][j] = vzero();

  for (int k0 = 0; k0 < D_; k0 += GK) {
    __syncthreads();
#pragma unroll
    for (int it = 0; it < 4; ++it)                       // GM = 4*32 rows
      *(float4*)(os + it * 32 * XSTR) =
          *(const float4*)(og + (size_t)it * 32 * D_ + k0);
#pragma unroll
    for (int it = 0; it < 4; ++it)                       // GN = 4*16 rows
      store_bf4(ws + it * 16 * XSTR,
                *(const float4*)(wg + (size_t)it * 16 * D_ + k0));
    __syncthreads();
    v16bf a0 = load_a_frag(Os + (wave * 32 +  0) * XSTR, XSTR, 0);
    v16bf a1 = load_a_frag(Os + (wave * 32 + 16) * XSTR, XSTR, 0);
#pragma unroll
    for (int j = 0; j < 4; ++j) {
      v16bf b = load_b_frag(Ws + j * 16 * XSTR, XSTR, 0);
      acc[0][j] = wmma_bf16(a0, b, acc[0][j]);
      acc[1][j] = wmma_bf16(a1, b, acc[1][j]);
    }
  }

  const int half  = lane >> 4;
  const int colLo = lane & 15;
#pragma unroll
  for (int mi = 0; mi < 2; ++mi) {
#pragma unroll
    for (int j = 0; j < 4; ++j) {
#pragma unroll
      for (int r = 0; r < 8; ++r) {
        const int m = m0 + wave * 32 + mi * 16 + r + (half << 3);
        out[(size_t)m * D_ + n0 + j * 16 + colLo] = acc[mi][j][r];
      }
    }
  }
}

// ---------------- host launcher ----------------
extern "C" void kernel_launch(void* const* d_in, const int* in_sizes, int n_in,
                              void* d_out, int out_size, void* d_ws, size_t ws_size,
                              hipStream_t stream) {
  const float* x    = (const float*)d_in[0];
  const float* wqkv = (const float*)d_in[1];
  const float* wout = (const float*)d_in[2];
  const float* cosT = (const float*)d_in[3];
  const float* sinT = (const float*)d_in[4];
  float* out = (float*)d_out;

  const size_t nelem = (size_t)B_ * S_ * D_;   // 8M elems per buffer
  char* ws = (char*)d_ws;
  __bf16* qw = (__bf16*)(ws + 0 * nelem * 2);
  __bf16* kw = (__bf16*)(ws + 1 * nelem * 2);
  __bf16* vw = (__bf16*)(ws + 2 * nelem * 2);
  __bf16* ow = (__bf16*)(ws + 3 * nelem * 2);

  dim3 blk(128);
  dim3 g1(3 * D_ / GN, (B_ * S_) / GM);   // 96 x 32
  qkv_rope_kernel<<<g1, blk, 0, stream>>>(x, wqkv, cosT, sinT, qw, kw, vw);
  dim3 g2(S_ / 64, B_ * H_);              // 32 x 32
  attn_kernel<<<g2, blk, 0, stream>>>(qw, kw, vw, ow);
  dim3 g3(D_ / GN, (B_ * S_) / GM);       // 32 x 32
  outproj_kernel<<<g3, blk, 0, stream>>>(ow, wout, out);
}
